// Encoder_36876589204306
// MI455X (gfx1250) — matrix operand order repaired and checked by reading, not compiled, not run on hardware
//
#include <hip/hip_runtime.h>

#define NB 64
#define NS 100
#define ND 1024
#define NH 16
#define NF 4096
#define NDEPTH 64
#define NBS (NB * NS)  // 6400

typedef __attribute__((ext_vector_type(16))) __bf16 bf16x16;
typedef __attribute__((ext_vector_type(8)))  float  f32x8;
typedef __attribute__((ext_vector_type(4)))  unsigned int u32x4;

union Frag { u32x4 q[2]; bf16x16 v; };

__device__ __forceinline__ unsigned short f2bf(float f) {
  union { float f; unsigned int u; } x; x.f = f;
  unsigned int r = 0x7FFFu + ((x.u >> 16) & 1u);
  return (unsigned short)((x.u + r) >> 16);
}

// ---- async global->LDS copy (CDNA5 ASYNCcnt path), with sync fallback ----
#if defined(__gfx1250__) && __has_builtin(__builtin_amdgcn_global_load_async_to_lds_b128)
#define HAVE_ASYNC 1
#else
#define HAVE_ASYNC 0
#endif

typedef int v4i_gcc __attribute__((vector_size(16)));
typedef __attribute__((address_space(1))) v4i_gcc as1_v4i;
typedef __attribute__((address_space(3))) v4i_gcc as3_v4i;

__device__ __forceinline__ void cp16(void* lds_dst, const void* g_src) {
#if HAVE_ASYNC
  // flat shared address low 32 bits == LDS byte offset (ISA 10.2 aperture rule)
  __builtin_amdgcn_global_load_async_to_lds_b128(
      (as1_v4i*)(unsigned long long)g_src,
      (as3_v4i*)(unsigned)(unsigned long long)lds_dst,
      0, 0);
#else
  *(u32x4*)lds_dst = *(const u32x4*)g_src;
#endif
}

__device__ __forceinline__ void wait_async() {
#if HAVE_ASYNC
#if __has_builtin(__builtin_amdgcn_s_wait_asynccnt)
  __builtin_amdgcn_s_wait_asynccnt(0);
#else
  asm volatile("s_wait_asynccnt 0" ::: "memory");
#endif
#endif
}

// ---------------- conversion / elementwise kernels ----------------

__global__ void __launch_bounds__(256)
cvt_f32_bf16(const float* __restrict__ in, unsigned short* __restrict__ out, long n) {
  long i = (long)blockIdx.x * 256 + threadIdx.x;
  if (i < n) out[i] = f2bf(in[i]);
}

// Wq/Wk/Wv [H, D, 64] -> bf16 [D, 3*1024] with column = which*1024 + h*64 + e
__global__ void __launch_bounds__(256)
pack_qkv_w(const float* __restrict__ Wq, const float* __restrict__ Wk,
           const float* __restrict__ Wv, unsigned short* __restrict__ out) {
  long i = (long)blockIdx.x * 256 + threadIdx.x;  // 3 * 2^20
  if (i >= 3L * ND * ND) return;
  int which = (int)(i >> 20);
  int rem = (int)(i & ((1 << 20) - 1));  // h*65536 + d*64 + e
  const float* W = which == 0 ? Wq : (which == 1 ? Wk : Wv);
  int h = rem >> 16;
  int e = rem & 63;
  int d = (rem >> 6) & 1023;
  int col = which * ND + h * NDEPTH + e;
  out[(long)d * (3 * ND) + col] = f2bf(W[rem]);
}

__global__ void __launch_bounds__(256)
pack_qkv_b(const float* bq, const float* bk, const float* bv, float* out) {
  int i = blockIdx.x * 256 + threadIdx.x;
  if (i >= 3 * ND) return;
  int which = i >> 10, j = i & 1023;
  out[i] = which == 0 ? bq[j] : (which == 1 ? bk[j] : bv[j]);
}

// x = emb_table[tokens] + PE, in f32 and bf16
__global__ void __launch_bounds__(256)
embed_pe(const int* __restrict__ tokens, const float* __restrict__ emb,
         float* __restrict__ x, unsigned short* __restrict__ xbf) {
  long i = (long)blockIdx.x * 256 + threadIdx.x;
  if (i >= (long)NBS * ND) return;
  int d = (int)(i & (ND - 1));
  long bs = i >> 10;
  int s = (int)(bs % NS);
  int tok = tokens[bs];
  float expo = (float)(d & ~1) * (1.0f / ND);
  float inv = __expf(-expo * 9.210340371976184f);  // 10000^(-expo)
  float ang = (float)s * inv;
  float pe = (d & 1) ? __cosf(ang) : __sinf(ang);
  float v = emb[(long)tok * ND + d] + pe;
  x[i] = v;
  xbf[i] = f2bf(v);
}

// ---------------- 128x128 bf16 WMMA GEMM, double-buffered async tiles -------
// A [M,K] bf16 row-major, B [K,N] bf16 row-major, C = A*B + bias
#define LDA_T 48    // 128 x (32+16) bf16, 16B-multiple row stride
#define LDB_T 144   // 32 x (128+16) bf16

template <int RELU, int WF32, int WBF16>
__global__ void __launch_bounds__(256)
gemm_bf16(const unsigned short* __restrict__ A, const unsigned short* __restrict__ Bm,
          const float* __restrict__ bias, float* __restrict__ Cf,
          unsigned short* __restrict__ Cb, int M, int N, int K) {
  __shared__ alignas(16) unsigned short Asl[2][128 * LDA_T];
  __shared__ alignas(16) unsigned short Bsl[2][32 * LDB_T];
  const int tid = threadIdx.x;
  const int lane = tid & 31;
  const int wid = tid >> 5;
  const int wm = wid & 3;   // wave M index (4)
  const int wn = wid >> 2;  // wave N index (2)
  const int half = lane >> 4;
  const int l16 = lane & 15;
  const int bm = blockIdx.y * 128;
  const int bn = blockIdx.x * 128;

  // per-thread tile-copy coordinates (fixed across K steps)
  const int ar0 = tid >> 2, ac0 = (tid & 3) * 8;
  const int ar1 = (tid + 256) >> 2, ac1 = ((tid + 256) & 3) * 8;
  const int br0 = tid >> 4, bc0 = (tid & 15) * 8;
  const int br1 = (tid + 256) >> 4, bc1 = ((tid + 256) & 15) * 8;

  auto issue_tile = [&](int k0, int buf) {
    cp16(&Asl[buf][ar0 * LDA_T + ac0], A + (long)(bm + ar0) * K + k0 + ac0);
    cp16(&Asl[buf][ar1 * LDA_T + ac1], A + (long)(bm + ar1) * K + k0 + ac1);
    cp16(&Bsl[buf][br0 * LDB_T + bc0], Bm + (long)(k0 + br0) * N + bn + bc0);
    cp16(&Bsl[buf][br1 * LDB_T + bc1], Bm + (long)(k0 + br1) * N + bn + bc1);
  };

  f32x8 acc[2][4];
#pragma unroll
  for (int mi = 0; mi < 2; ++mi)
#pragma unroll
    for (int ni = 0; ni < 4; ++ni)
#pragma unroll
      for (int r = 0; r < 8; ++r) acc[mi][ni][r] = 0.0f;

  issue_tile(0, 0);
  wait_async();
  __syncthreads();

  for (int k0 = 0; k0 < K; k0 += 32) {
    const int cur = (k0 >> 5) & 1;
    if (k0 + 32 < K) issue_tile(k0 + 32, cur ^ 1);  // overlap copy with WMMA

    Frag af[2], bf[4];
#pragma unroll
    for (int mi = 0; mi < 2; ++mi) {
      // A frag: lane row = M, K pairs; halves offset by 8 in K
      const unsigned short* p = &Asl[cur][(wm * 32 + mi * 16 + l16) * LDA_T + half * 8];
      af[mi].q[0] = *(const u32x4*)(p);
      af[mi].q[1] = *(const u32x4*)(p + 16);
    }
#pragma unroll
    for (int ni = 0; ni < 4; ++ni) {
      // B frag: lane column = N, K rows in table order (halves offset by 8 in K)
      const unsigned short* p = &Bsl[cur][(half * 8) * LDB_T + wn * 64 + ni * 16 + l16];
      for (int j = 0; j < 2; ++j)
        for (int i = 0; i < 4; ++i)
          bf[ni].q[j][i] = (unsigned)p[(j * 16 + 2 * i) * LDB_T] | ((unsigned)p[(j * 16 + 2 * i + 1) * LDB_T] << 16);
    }
#pragma unroll
    for (int mi = 0; mi < 2; ++mi)
#pragma unroll
      for (int ni = 0; ni < 4; ++ni)
        acc[mi][ni] = __builtin_amdgcn_wmma_f32_16x16x32_bf16(
            false, af[mi].v, false, bf[ni].v, (short)0, acc[mi][ni], false, false);

    wait_async();
    __syncthreads();
  }

#pragma unroll
  for (int mi = 0; mi < 2; ++mi)
#pragma unroll
    for (int ni = 0; ni < 4; ++ni) {
      int col = bn + wn * 64 + ni * 16 + l16;
      float bv = bias[col];
#pragma unroll
      for (int r = 0; r < 8; ++r) {
        int row = bm + wm * 32 + mi * 16 + r + 8 * half;
        float v = acc[mi][ni][r] + bv;
        if (RELU) v = v > 0.f ? v : 0.f;
        long o = (long)row * N + col;
        if (WF32) Cf[o] = v;
        if (WBF16) Cb[o] = f2bf(v);
      }
    }
}

// ---------------- attention: one workgroup per (b, h) ----------------
#define SQ 112    // padded S (7 x 16)
#define SKK 128   // padded K-dim for P@V (4 x 32)
#define LQ 72     // Q lds stride (elems)
#define LKT 120   // K^T lds stride
#define LV 72     // V lds stride
#define LSC 116   // f32 scores stride
#define LP 136    // P (probabilities) stride

#define OFF_KT (SQ * LQ * 2)            // 16128
#define OFF_V  (OFF_KT + 64 * LKT * 2)  // 31488
#define OFF_SC (OFF_V + SKK * LV * 2)   // 49920
#define SMEM_BYTES (OFF_SC + SQ * LSC * 4)  // 101888 (P aliases Q+KT region)

__global__ void __launch_bounds__(256)
attention(const unsigned short* __restrict__ qkv, unsigned short* __restrict__ ctx) {
  __shared__ alignas(16) char smem[SMEM_BYTES];
  unsigned short* Qs  = (unsigned short*)(smem);
  unsigned short* KTs = (unsigned short*)(smem + OFF_KT);
  unsigned short* Vs  = (unsigned short*)(smem + OFF_V);
  unsigned short* Ps  = (unsigned short*)(smem);  // aliases Q/KT after scores done
  float* Sc = (float*)(smem + OFF_SC);

  const int h = blockIdx.x;
  const int b = blockIdx.y;
  const int tid = threadIdx.x;
  const int lane = tid & 31;
  const int wid = tid >> 5;
  const int half = lane >> 4;
  const int l16 = lane & 15;
  const long base = (long)b * NS * (3 * ND) + (long)h * NDEPTH;

  // Q tile [112 x 64], rows >= 100 zero
  for (int idx = tid; idx < SQ * 8; idx += 256) {
    int r = idx >> 3, c = (idx & 7) * 8;
    u32x4 v = {0u, 0u, 0u, 0u};
    if (r < NS) v = *(const u32x4*)(qkv + base + (long)r * (3 * ND) + c);
    *(u32x4*)(Qs + r * LQ + c) = v;
  }
  // V tile [128 x 64]
  for (int idx = tid; idx < SKK * 8; idx += 256) {
    int r = idx >> 3, c = (idx & 7) * 8;
    u32x4 v = {0u, 0u, 0u, 0u};
    if (r < NS) v = *(const u32x4*)(qkv + base + 2 * ND + (long)r * (3 * ND) + c);
    *(u32x4*)(Vs + r * LV + c) = v;
  }
  // K^T tile [64 x 112] (transpose on the fly)
  for (int idx = tid; idx < 64 * SQ; idx += 256) {
    int e = idx & 63, s = idx >> 6;
    unsigned short v = 0;
    if (s < NS) v = qkv[base + ND + (long)s * (3 * ND) + e];
    KTs[e * LKT + s] = v;
  }
  __syncthreads();

  // scores = (Q K^T) / 8 : 7x7 tiles, K-dim = 64
  for (int t = wid; t < 49; t += 8) {
    int mt = t / 7, nt = t % 7;
    f32x8 a;
#pragma unroll
    for (int r = 0; r < 8; ++r) a[r] = 0.f;
#pragma unroll
    for (int ks = 0; ks < 2; ++ks) {
      Frag af, bf;
      const unsigned short* pA = Qs + (mt * 16 + l16) * LQ + ks * 32 + half * 8;
      af.q[0] = *(const u32x4*)(pA);
      af.q[1] = *(const u32x4*)(pA + 16);
      const unsigned short* pB = KTs + (ks * 32 + half * 8) * LKT + nt * 16 + l16;
      for (int j = 0; j < 2; ++j)
        for (int i = 0; i < 4; ++i)
          bf.q[j][i] = (unsigned)pB[(j * 16 + 2 * i) * LKT] | ((unsigned)pB[(j * 16 + 2 * i + 1) * LKT] << 16);
      a = __builtin_amdgcn_wmma_f32_16x16x32_bf16(false, af.v, false, bf.v, (short)0, a,
                                                  false, false);
    }
#pragma unroll
    for (int r = 0; r < 8; ++r)
      Sc[(mt * 16 + r + 8 * half) * LSC + nt * 16 + l16] = a[r] * 0.125f;
  }
  __syncthreads();

  // row softmax over the 100 valid keys; write bf16 P (cols 100..127 = 0)
  for (int row = tid; row < SQ; row += 256) {
    float mx = -1e30f;
    for (int c = 0; c < NS; ++c) { float v = Sc[row * LSC + c]; mx = v > mx ? v : mx; }
    float sum = 0.f;
    for (int c = 0; c < NS; ++c) {
      float e = __expf(Sc[row * LSC + c] - mx);
      Sc[row * LSC + c] = e;
      sum += e;
    }
    float inv = 1.0f / sum;
    for (int c = 0; c < NS; ++c) Ps[row * LP + c] = f2bf(Sc[row * LSC + c] * inv);
    for (int c = NS; c < SKK; ++c) Ps[row * LP + c] = 0;
  }
  __syncthreads();

  // ctx = P @ V : 7x4 tiles, K-dim = 128
  for (int t = wid; t < 28; t += 8) {
    int mt = t >> 2, nt = t & 3;
    f32x8 a;
#pragma unroll
    for (int r = 0; r < 8; ++r) a[r] = 0.f;
#pragma unroll
    for (int ks = 0; ks < 4; ++ks) {
      Frag af, bf;
      const unsigned short* pA = Ps + (mt * 16 + l16) * LP + ks * 32 + half * 8;
      af.q[0] = *(const u32x4*)(pA);
      af.q[1] = *(const u32x4*)(pA + 16);
      const unsigned short* pB = Vs + (ks * 32 + half * 8) * LV + nt * 16 + l16;
      for (int j = 0; j < 2; ++j)
        for (int i = 0; i < 4; ++i)
          bf.q[j][i] = (unsigned)pB[(j * 16 + 2 * i) * LV] | ((unsigned)pB[(j * 16 + 2 * i + 1) * LV] << 16);
      a = __builtin_amdgcn_wmma_f32_16x16x32_bf16(false, af.v, false, bf.v, (short)0, a,
                                                  false, false);
    }
#pragma unroll
    for (int r = 0; r < 8; ++r) {
      int row = mt * 16 + r + 8 * half;
      if (row < NS)
        ctx[((long)b * NS + row) * ND + h * NDEPTH + nt * 16 + l16] = f2bf(a[r]);
    }
  }
}

// ---------------- residual add + LayerNorm (row = 1024) ----------------
template <int WBF16>
__global__ void __launch_bounds__(256)
add_ln(const float* __restrict__ a, const float* __restrict__ res,
       const float* __restrict__ gamma, const float* __restrict__ beta,
       float* __restrict__ outf, unsigned short* __restrict__ outb) {
  __shared__ float red[256];
  const int row = blockIdx.x;
  const int tid = threadIdx.x;
  const long o = (long)row * ND;
  float t[4];
  float s = 0.f;
#pragma unroll
  for (int j = 0; j < 4; ++j) {
    int c = tid + j * 256;
    t[j] = a[o + c] + res[o + c];
    s += t[j];
  }
  red[tid] = s;
  __syncthreads();
  for (int k = 128; k > 0; k >>= 1) { if (tid < k) red[tid] += red[tid + k]; __syncthreads(); }
  float mean = red[0] * (1.0f / ND);
  __syncthreads();
  float s2 = 0.f;
#pragma unroll
  for (int j = 0; j < 4; ++j) { float d = t[j] - mean; s2 += d * d; }
  red[tid] = s2;
  __syncthreads();
  for (int k = 128; k > 0; k >>= 1) { if (tid < k) red[tid] += red[tid + k]; __syncthreads(); }
  float rstd = rsqrtf(red[0] * (1.0f / ND) + 1e-6f);
#pragma unroll
  for (int j = 0; j < 4; ++j) {
    int c = tid + j * 256;
    float y = gamma[c] * (t[j] - mean) * rstd + beta[c];
    outf[o + c] = y;
    if (WBF16) outb[o + c] = f2bf(y);
  }
}

// ---------------- host launcher ----------------
extern "C" void kernel_launch(void* const* d_in, const int* in_sizes, int n_in,
                              void* d_out, int out_size, void* d_ws, size_t ws_size,
                              hipStream_t stream) {
  const int* tokens = (const int*)d_in[0];
  const float* emb = (const float*)d_in[1];
  const float* Wq = (const float*)d_in[2];
  const float* bq = (const float*)d_in[3];
  const float* Wk = (const float*)d_in[4];
  const float* bk = (const float*)d_in[5];
  const float* Wv = (const float*)d_in[6];
  const float* bv = (const float*)d_in[7];
  const float* Wo = (const float*)d_in[8];
  const float* bo = (const float*)d_in[9];
  const float* W1 = (const float*)d_in[10];
  const float* b1 = (const float*)d_in[11];
  const float* W2 = (const float*)d_in[12];
  const float* b2 = (const float*)d_in[13];
  const float* g1 = (const float*)d_in[14];
  const float* be1 = (const float*)d_in[15];
  const float* g2 = (const float*)d_in[16];
  const float* be2 = (const float*)d_in[17];
  (void)in_sizes; (void)n_in; (void)out_size; (void)ws_size;

  char* ws = (char*)d_ws;
  size_t off = 0;
  auto take = [&](size_t bytes) {
    char* p = ws + off;
    off += (bytes + 255) & ~(size_t)255;
    return p;
  };

  float* x             = (float*)take((size_t)NBS * ND * 4);
  unsigned short* xbf  = (unsigned short*)take((size_t)NBS * ND * 2);
  unsigned short* wqkv = (unsigned short*)take((size_t)ND * 3 * ND * 2);
  float* bqkv          = (float*)take((size_t)3 * ND * 4);
  unsigned short* wob  = (unsigned short*)take((size_t)ND * ND * 2);
  unsigned short* w1b  = (unsigned short*)take((size_t)ND * NF * 2);
  unsigned short* w2b  = (unsigned short*)take((size_t)NF * ND * 2);
  unsigned short* qkv  = (unsigned short*)take((size_t)NBS * 3 * ND * 2);
  unsigned short* ctx  = (unsigned short*)take((size_t)NBS * ND * 2);
  float* attn_out      = (float*)take((size_t)NBS * ND * 4);
  float* x1            = (float*)take((size_t)NBS * ND * 4);
  unsigned short* x1b  = (unsigned short*)take((size_t)NBS * ND * 2);
  unsigned short* hid  = (unsigned short*)take((size_t)NBS * NF * 2);
  float* ff            = (float*)take((size_t)NBS * ND * 4);

  // weight / bias conversion to bf16
  pack_qkv_w<<<(3 * ND * ND + 255) / 256, 256, 0, stream>>>(Wq, Wk, Wv, wqkv);
  pack_qkv_b<<<(3 * ND + 255) / 256, 256, 0, stream>>>(bq, bk, bv, bqkv);
  cvt_f32_bf16<<<(ND * ND + 255) / 256, 256, 0, stream>>>(Wo, wob, (long)ND * ND);
  cvt_f32_bf16<<<(ND * NF + 255) / 256, 256, 0, stream>>>(W1, w1b, (long)ND * NF);
  cvt_f32_bf16<<<(ND * NF + 255) / 256, 256, 0, stream>>>(W2, w2b, (long)NF * ND);

  // x = embed + positional encoding
  embed_pe<<<(NBS * ND + 255) / 256, 256, 0, stream>>>(tokens, emb, x, xbf);

  // fused QKV projection: [6400,1024] x [1024,3072]
  gemm_bf16<0, 0, 1><<<dim3(3 * ND / 128, NBS / 128), 256, 0, stream>>>(
      xbf, wqkv, bqkv, nullptr, qkv, NBS, 3 * ND, ND);

  // attention per (b, h)
  attention<<<dim3(NH, NB), 256, 0, stream>>>(qkv, ctx);

  // output projection: [6400,1024] x [1024,1024]
  gemm_bf16<0, 1, 0><<<dim3(ND / 128, NBS / 128), 256, 0, stream>>>(
      ctx, wob, bo, attn_out, nullptr, NBS, ND, ND);

  // x1 = LN(attn_out + x)
  add_ln<1><<<NBS, 256, 0, stream>>>(attn_out, x, g1, be1, x1, x1b);

  // FFN: relu(x1 @ W1 + b1) @ W2 + b2
  gemm_bf16<1, 0, 1><<<dim3(NF / 128, NBS / 128), 256, 0, stream>>>(
      x1b, w1b, b1, nullptr, hid, NBS, NF, ND);
  gemm_bf16<0, 1, 0><<<dim3(ND / 128, NBS / 128), 256, 0, stream>>>(
      hid, w2b, b2, ff, nullptr, NBS, ND, NF);

  // out = LN(ff + x1)
  add_ln<0><<<NBS, 256, 0, stream>>>(ff, x1, g2, be2, (float*)d_out, nullptr);
}
